// AttentionModel_73778948210806
// MI455X (gfx1250) — matrix operand order repaired and checked by reading, not compiled, not run on hardware
//
#include <hip/hip_runtime.h>

// ---------------------------------------------------------------------------
// Problem constants (from reference)
// ---------------------------------------------------------------------------
#define BB   256          // batch
#define NN   100          // nodes
#define EE   128          // embedding
#define MH   8            // heads
#define DK   16           // head dim
#define DFF  512          // ffn hidden
#define NL   3            // layers
#define BN   (BB * NN)    // 25600 rows
#define EPSV 1e-5f

// ---------------------------------------------------------------------------
// WMMA / TDM types (CDNA5 / gfx1250, wave32)
// ---------------------------------------------------------------------------
typedef __attribute__((ext_vector_type(16))) __bf16 v16bf;
typedef __attribute__((ext_vector_type(8)))  __bf16 v8bf;
typedef __attribute__((ext_vector_type(8)))  float  v8f;
typedef __attribute__((ext_vector_type(4)))  unsigned int v4u;
typedef __attribute__((ext_vector_type(8)))  int v8i;
typedef __attribute__((ext_vector_type(4)))  int v4i;

__device__ __forceinline__ v16bf make16(v8bf lo, v8bf hi) {
    v16bf r;
#pragma unroll
    for (int i = 0; i < 8; ++i) { r[i] = lo[i]; r[i + 8] = hi[i]; }
    return r;
}

// A fragment: 16x32 (MxK) bf16. Lane l: m = mrow + (l&15); K-halves split by l>>4.
// elements 0..7  <- A[m][kbase + (l>>4)*8 + 0..7]
// elements 8..15 <- A[m][kbase + 16 + (l>>4)*8 + 0..7]
__device__ __forceinline__ v16bf load_a_frag(const __bf16* A, int lda,
                                             int mrow, int kbase, int lane) {
    int m  = mrow + (lane & 15);
    int kg = (lane >> 4) * 8;
    const __bf16* p0 = A + (size_t)m * lda + kbase + kg;
    const __bf16* p1 = A + (size_t)m * lda + kbase + 16 + kg;
    return make16(*(const v8bf*)p0, *(const v8bf*)p1);
}

// ---------------------------------------------------------------------------
// Generic GEMM: C[M,N] = A[M,K](bf16, row-major) * Bw[K,N](bf16, row-major) + bias
// Optional relu. Writes fp32 and/or bf16 outputs (either may be null).
// Block: 256 thr = 8 waves arranged 4(M) x 2(N); wave tile = 16M x 32N;
// block tile = 64M x 64N. Requires M%64==0, N%64==0, K%32==0 (true here).
//
// The block's B panel (K x 64 bf16, <= 64 KB) is DMA'd into LDS once via the
// Tensor Data Mover; all 4 M-waves then read B fragments from LDS (4x reuse,
// no redundant global traffic). A is streamed with global_load_b128 so the
// vector loads overlap the WMMA pipe via LOADcnt.
// ---------------------------------------------------------------------------
__global__ __launch_bounds__(256)
void gemm_bf16_kernel(const __bf16* __restrict__ A, const __bf16* __restrict__ Bw,
                      const float* __restrict__ bias,
                      float* __restrict__ Cf, __bf16* __restrict__ Cb,
                      int Mtot, int Ntot, int K, int relu) {
    extern __shared__ __align__(16) char smem[];
    __bf16* sB = (__bf16*)smem;              // K x 64 bf16, row stride 64

    const int lane = threadIdx.x & 31;
    const int wave = threadIdx.x >> 5;       // 0..7
    const int wm   = wave >> 1;              // 0..3
    const int wn   = wave & 1;               // 0..1
    const int mrow = blockIdx.x * 64 + wm * 16;
    const int nblk = blockIdx.y * 64;
    const int nloc = wn * 32;

    // ---- TDM: load B panel (rows 0..K-1, cols nblk..nblk+63) into LDS ----
    if (wave == 0) {
        unsigned lds_base = (unsigned)(size_t)(void*)sB;
        unsigned long long ga = (unsigned long long)(size_t)(Bw + nblk);
        const unsigned td0 = (unsigned)Ntot;   // tensor dim0 (elements)
        const unsigned td1 = (unsigned)K;      // tensor dim1 (rows)
        const unsigned st0 = (unsigned)Ntot;   // dim0 stride (elements)

        v4u g0;
        g0[0] = 1u;                                            // count=1, user mode
        g0[1] = lds_base;                                      // lds_addr
        g0[2] = (unsigned)(ga & 0xFFFFFFFFu);                  // global_addr[31:0]
        g0[3] = (unsigned)((ga >> 32) & 0x01FFFFFFu) | (2u << 30); // [56:32] | type=2

        v8i g1;
        g1[0] = (int)(1u << 16);                               // data_size=1 (2B)
        g1[1] = (int)((td0 & 0xFFFFu) << 16);                  // tensor_dim0[15:0]
        g1[2] = (int)(((td0 >> 16) & 0xFFFFu) | ((td1 & 0xFFFFu) << 16));
        g1[3] = (int)(((td1 >> 16) & 0xFFFFu) | (64u << 16));  // tile_dim0=64
        g1[4] = (int)(td1 & 0xFFFFu);                          // tile_dim1=K, tile_dim2=0
        g1[5] = (int)st0;                                      // dim0_stride[31:0]
        g1[6] = 0;                                             // stride hi / dim1_stride
        g1[7] = 0;

        v4i gz4 = {0, 0, 0, 0};                                // 2D tile: groups 2/3 unused
        v8i gz8 = {0, 0, 0, 0, 0, 0, 0, 0};
        __builtin_amdgcn_tensor_load_to_lds(g0, g1, gz4, gz4, gz8, 0);
        __builtin_amdgcn_s_wait_tensorcnt(0);
    }
    __syncthreads();

    // ---- K loop: A from global, B fragments from LDS ----
    v8f acc0 = {};
    v8f acc1 = {};
    for (int k0 = 0; k0 < K; k0 += 32) {
        v16bf a = load_a_frag(A, K, mrow, k0, lane);
        // B fragment: lane l -> n = nloc + (l&15); element s -> k = k0 + (s<8 ? 8h+s : 16+8h+s-8)
        const __bf16* bp = sB + nloc + (lane & 15);
        v16bf b0, b1;
#pragma unroll
        for (int s = 0; s < 16; ++s) {
            int kk = k0 + (lane >> 4) * 8 + (s < 8 ? s : 8 + s);
            b0[s] = bp[(size_t)kk * 64];
            b1[s] = bp[(size_t)kk * 64 + 16];
        }
        acc0 = __builtin_amdgcn_wmma_f32_16x16x32_bf16(false, a, false, b0,
                                                       (short)0, acc0, false, false);
        acc1 = __builtin_amdgcn_wmma_f32_16x16x32_bf16(false, a, false, b1,
                                                       (short)0, acc1, false, false);
    }

    // C/D layout: lane -> n = (lane&15); element e -> m = mrow + 8*(lane>>4) + e
    const int n0 = nblk + nloc + (lane & 15);
    const int n1 = n0 + 16;
    const int mb = mrow + 8 * (lane >> 4);
    const float bia0 = bias[n0];
    const float bia1 = bias[n1];
#pragma unroll
    for (int e = 0; e < 8; ++e) {
        const int m = mb + e;
        float v0 = acc0[e] + bia0;
        float v1 = acc1[e] + bia1;
        if (relu) { v0 = fmaxf(v0, 0.0f); v1 = fmaxf(v1, 0.0f); }
        const size_t i0 = (size_t)m * Ntot + n0;
        const size_t i1 = (size_t)m * Ntot + n1;
        if (Cf) { Cf[i0] = v0; Cf[i1] = v1; }
        if (Cb) { Cb[i0] = (__bf16)v0; Cb[i1] = (__bf16)v1; }
    }
}

// ---------------------------------------------------------------------------
// Embedding: node = [s, d] @ emb_w + emb_b ; node 0 = s0 @ embp_w + embp_b
// ---------------------------------------------------------------------------
__global__ __launch_bounds__(256)
void embed_kernel(const float* __restrict__ s, const int* __restrict__ d,
                  const float* __restrict__ ew, const float* __restrict__ eb,
                  const float* __restrict__ pw, const float* __restrict__ pb,
                  float* __restrict__ xw, __bf16* __restrict__ xb) {
    int idx = blockIdx.x * blockDim.x + threadIdx.x;
    if (idx >= BN * EE) return;
    int e  = idx & (EE - 1);
    int bn = idx >> 7;
    int n  = bn % NN;
    float s0 = s[(size_t)bn * 2 + 0];
    float s1 = s[(size_t)bn * 2 + 1];
    float val;
    if (n == 0) {
        val = s0 * pw[0 * EE + e] + s1 * pw[1 * EE + e] + pb[e];
    } else {
        float dd = (float)d[bn];
        val = s0 * ew[0 * EE + e] + s1 * ew[1 * EE + e] + dd * ew[2 * EE + e] + eb[e];
    }
    xw[idx] = val;
    xb[idx] = (__bf16)val;
}

// ---------------------------------------------------------------------------
// Attention core per (b, head): scores = qk^T/4, softmax over keys, z = attn @ v.
// K/V staged in LDS; one thread per query row; 2-pass softmax (max, then
// exp + weighted-V accumulation). Writes bf16 z for the wo GEMM.
// ---------------------------------------------------------------------------
__global__ __launch_bounds__(128)
void attn_kernel(const float* __restrict__ q, const float* __restrict__ k,
                 const float* __restrict__ v, __bf16* __restrict__ zb) {
    __shared__ float sk[NN * DK];
    __shared__ float sv[NN * DK];
    const int bm = blockIdx.x;            // 0 .. B*MH-1
    const int b  = bm >> 3;
    const int m  = bm & (MH - 1);
    const size_t base = (size_t)b * NN * EE + (size_t)m * DK;

    for (int t = threadIdx.x; t < NN * DK; t += blockDim.x) {
        int j = t >> 4, dd = t & 15;
        sk[t] = k[base + (size_t)j * EE + dd];
        sv[t] = v[base + (size_t)j * EE + dd];
    }
    __syncthreads();

    const int i = threadIdx.x;
    if (i < NN) {
        float qi[DK];
#pragma unroll
        for (int dd = 0; dd < DK; ++dd) qi[dd] = q[base + (size_t)i * EE + dd];

        float mx = -3.0e38f;
        for (int j = 0; j < NN; ++j) {
            float sc = 0.0f;
#pragma unroll
            for (int dd = 0; dd < DK; ++dd) sc += qi[dd] * sk[j * DK + dd];
            mx = fmaxf(mx, sc * 0.25f);
        }
        float z[DK];
#pragma unroll
        for (int dd = 0; dd < DK; ++dd) z[dd] = 0.0f;
        float sum = 0.0f;
        for (int j = 0; j < NN; ++j) {
            float sc = 0.0f;
#pragma unroll
            for (int dd = 0; dd < DK; ++dd) sc += qi[dd] * sk[j * DK + dd];
            float p = __expf(sc * 0.25f - mx);
            sum += p;
#pragma unroll
            for (int dd = 0; dd < DK; ++dd) z[dd] += p * sv[j * DK + dd];
        }
        float inv = 1.0f / sum;
#pragma unroll
        for (int dd = 0; dd < DK; ++dd)
            zb[base + (size_t)i * EE + dd] = (__bf16)(z[dd] * inv);
    }
}

// ---------------------------------------------------------------------------
// Residual add (in place): x += y
// ---------------------------------------------------------------------------
__global__ __launch_bounds__(256)
void add_kernel(float* __restrict__ x, const float* __restrict__ y, int n) {
    int i = blockIdx.x * blockDim.x + threadIdx.x;
    if (i < n) x[i] += y[i];
}

// ---------------------------------------------------------------------------
// BatchNorm (train mode over B*N rows, per channel), deterministic 2-stage.
// ---------------------------------------------------------------------------
__global__ __launch_bounds__(256)
void bn_stats_kernel(const float* __restrict__ x,
                     float* __restrict__ psum, float* __restrict__ psumsq) {
    __shared__ float ssum[256];
    __shared__ float ssq[256];
    const int c    = threadIdx.x & (EE - 1);
    const int half = threadIdx.x >> 7;             // 0/1 -> even/odd rows
    const size_t row0 = (size_t)blockIdx.x * 128 + half;
    float s = 0.0f, q = 0.0f;
    for (int r = 0; r < 128; r += 2) {
        float vv = x[(row0 + r) * EE + c];
        s += vv;
        q += vv * vv;
    }
    ssum[threadIdx.x] = s;
    ssq[threadIdx.x]  = q;
    __syncthreads();
    if (half == 0) {
        psum[blockIdx.x * EE + c]   = ssum[c] + ssum[c + 128];
        psumsq[blockIdx.x * EE + c] = ssq[c] + ssq[c + 128];
    }
}

__global__ __launch_bounds__(128)
void bn_finalize_kernel(const float* __restrict__ psum, const float* __restrict__ psumsq,
                        int nblk, float* __restrict__ meanrstd) {
    const int c = threadIdx.x;                     // 128 channels
    float s = 0.0f, q = 0.0f;
    for (int b = 0; b < nblk; ++b) {
        s += psum[b * EE + c];
        q += psumsq[b * EE + c];
    }
    const float inv_n = 1.0f / (float)BN;
    float mean = s * inv_n;
    float var  = q * inv_n - mean * mean;
    meanrstd[c]      = mean;
    meanrstd[EE + c] = rsqrtf(var + EPSV);
}

__global__ __launch_bounds__(256)
void bn_apply_kernel(float* __restrict__ x, __bf16* __restrict__ xb,
                     const float* __restrict__ meanrstd,
                     const float* __restrict__ g, const float* __restrict__ bta) {
    int idx = blockIdx.x * blockDim.x + threadIdx.x;
    if (idx >= BN * EE) return;
    int c = idx & (EE - 1);
    float val = g[c] * (x[idx] - meanrstd[c]) * meanrstd[EE + c] + bta[c];
    x[idx]  = val;
    xb[idx] = (__bf16)val;
}

// ---------------------------------------------------------------------------
// f32 -> bf16 weight conversion
// ---------------------------------------------------------------------------
__global__ __launch_bounds__(256)
void f32_to_bf16_kernel(const float* __restrict__ src, __bf16* __restrict__ dst, int n) {
    int i = blockIdx.x * blockDim.x + threadIdx.x;
    if (i < n) dst[i] = (__bf16)src[i];
}

// ---------------------------------------------------------------------------
// Host-side orchestration
// ---------------------------------------------------------------------------
static inline void launch_gemm(const __bf16* A, const __bf16* Bw, const float* bias,
                               float* Cf, __bf16* Cb, int Mtot, int Ntot, int K,
                               int relu, hipStream_t stream) {
    dim3 grid(Mtot / 64, Ntot / 64);
    size_t shmem = (size_t)K * 64 * sizeof(__bf16);   // B panel in LDS (<= 64 KB)
    gemm_bf16_kernel<<<grid, 256, shmem, stream>>>(A, Bw, bias, Cf, Cb,
                                                   Mtot, Ntot, K, relu);
}

extern "C" void kernel_launch(void* const* d_in, const int* in_sizes, int n_in,
                              void* d_out, int out_size, void* d_ws, size_t ws_size,
                              hipStream_t stream) {
    // ---- inputs (setup_inputs order) ----
    const float* s      = (const float*)d_in[0];
    const int*   d      = (const int*)  d_in[1];
    const float* emb_w  = (const float*)d_in[2];
    const float* emb_b  = (const float*)d_in[3];
    const float* embp_w = (const float*)d_in[4];
    const float* embp_b = (const float*)d_in[5];
    const float* wq_w   = (const float*)d_in[6];
    const float* wq_b   = (const float*)d_in[7];
    const float* wk_w   = (const float*)d_in[8];
    const float* wk_b   = (const float*)d_in[9];
    const float* wv_w   = (const float*)d_in[10];
    const float* wv_b   = (const float*)d_in[11];
    const float* wo_w   = (const float*)d_in[12];
    const float* wo_b   = (const float*)d_in[13];
    const float* fw_w   = (const float*)d_in[14];
    const float* fw_b   = (const float*)d_in[15];
    const float* fb_w   = (const float*)d_in[16];
    const float* fb_b   = (const float*)d_in[17];
    const float* bn1_g  = (const float*)d_in[18];
    const float* bn1_b  = (const float*)d_in[19];
    const float* bn2_g  = (const float*)d_in[20];
    const float* bn2_b  = (const float*)d_in[21];
    (void)in_sizes; (void)n_in; (void)out_size; (void)ws_size;

    // ---- workspace layout (all offsets 256B aligned) ----
    const size_t szXf = (size_t)BN * EE * sizeof(float);   // 13.1 MB
    const size_t szXb = (size_t)BN * EE * sizeof(__bf16);  //  6.6 MB
    const size_t szHb = (size_t)BN * DFF * sizeof(__bf16); // 26.2 MB
    const int NBLK = BN / 128;                             // 200 stats blocks

    char* p = (char*)d_ws;
    float*  xw = (float*)p;               p += szXf;
    float*  y  = (float*)p;               p += szXf;
    float*  q  = (float*)p;               p += szXf;
    float*  k  = (float*)p;               p += szXf;
    float*  v  = (float*)p;               p += szXf;
    __bf16* xb = (__bf16*)p;              p += szXb;
    __bf16* zb = (__bf16*)p;              p += szXb;
    __bf16* hb = (__bf16*)p;              p += szHb;
    __bf16* wbf = (__bf16*)p;             p += (size_t)(4 * NL * EE * EE
                                                + 2 * NL * EE * DFF) * sizeof(__bf16);
    float* psum   = (float*)p;            p += (size_t)NBLK * EE * sizeof(float);
    float* psumsq = (float*)p;            p += (size_t)NBLK * EE * sizeof(float);
    float* meanrstd = (float*)p;          p += 2 * EE * sizeof(float);

    // bf16 weight slots
    __bf16* wq_bf = wbf;
    __bf16* wk_bf = wq_bf + (size_t)NL * EE * EE;
    __bf16* wv_bf = wk_bf + (size_t)NL * EE * EE;
    __bf16* wo_bf = wv_bf + (size_t)NL * EE * EE;
    __bf16* fw_bf = wo_bf + (size_t)NL * EE * EE;
    __bf16* fb_bf = fw_bf + (size_t)NL * EE * DFF;

    const int nEW  = NL * EE * EE;    // 49152
    const int nFW  = NL * EE * DFF;   // 196608
    f32_to_bf16_kernel<<<(nEW + 255) / 256, 256, 0, stream>>>(wq_w, wq_bf, nEW);
    f32_to_bf16_kernel<<<(nEW + 255) / 256, 256, 0, stream>>>(wk_w, wk_bf, nEW);
    f32_to_bf16_kernel<<<(nEW + 255) / 256, 256, 0, stream>>>(wv_w, wv_bf, nEW);
    f32_to_bf16_kernel<<<(nEW + 255) / 256, 256, 0, stream>>>(wo_w, wo_bf, nEW);
    f32_to_bf16_kernel<<<(nFW + 255) / 256, 256, 0, stream>>>(fw_w, fw_bf, nFW);
    f32_to_bf16_kernel<<<(nFW + 255) / 256, 256, 0, stream>>>(fb_w, fb_bf, nFW);

    // ---- embedding ----
    const int nElem = BN * EE;
    embed_kernel<<<(nElem + 255) / 256, 256, 0, stream>>>(
        s, d, emb_w, emb_b, embp_w, embp_b, xw, xb);

    // ---- encoder layers ----
    for (int l = 0; l < NL; ++l) {
        const __bf16* wqL = wq_bf + (size_t)l * EE * EE;
        const __bf16* wkL = wk_bf + (size_t)l * EE * EE;
        const __bf16* wvL = wv_bf + (size_t)l * EE * EE;
        const __bf16* woL = wo_bf + (size_t)l * EE * EE;
        const __bf16* fwL = fw_bf + (size_t)l * EE * DFF;
        const __bf16* fbL = fb_bf + (size_t)l * DFF * EE;

        // Q, K, V projections (WMMA GEMMs, fp32 out for attention)
        launch_gemm(xb, wqL, wq_b + l * EE, q, nullptr, BN, EE, EE, 0, stream);
        launch_gemm(xb, wkL, wk_b + l * EE, k, nullptr, BN, EE, EE, 0, stream);
        launch_gemm(xb, wvL, wv_b + l * EE, v, nullptr, BN, EE, EE, 0, stream);

        // attention per (batch, head)
        attn_kernel<<<BB * MH, 128, 0, stream>>>(q, k, v, zb);

        // output projection
        launch_gemm(zb, woL, wo_b + l * EE, y, nullptr, BN, EE, EE, 0, stream);

        // residual + BN1
        add_kernel<<<(nElem + 255) / 256, 256, 0, stream>>>(xw, y, nElem);
        bn_stats_kernel<<<NBLK, 256, 0, stream>>>(xw, psum, psumsq);
        bn_finalize_kernel<<<1, 128, 0, stream>>>(psum, psumsq, NBLK, meanrstd);
        bn_apply_kernel<<<(nElem + 255) / 256, 256, 0, stream>>>(
            xw, xb, meanrstd, bn1_g + l * EE, bn1_b + l * EE);

        // FFN: relu(x @ fw + b) @ fb + b
        launch_gemm(xb, fwL, fw_b + l * DFF, nullptr, hb, BN, DFF, EE, 1, stream);
        launch_gemm(hb, fbL, fb_b + l * EE, y, nullptr, BN, EE, DFF, 0, stream);

        // residual + BN2
        add_kernel<<<(nElem + 255) / 256, 256, 0, stream>>>(xw, y, nElem);
        bn_stats_kernel<<<NBLK, 256, 0, stream>>>(xw, psum, psumsq);
        bn_finalize_kernel<<<1, 128, 0, stream>>>(psum, psumsq, NBLK, meanrstd);
        bn_apply_kernel<<<(nElem + 255) / 256, 256, 0, stream>>>(
            xw, xb, meanrstd, bn2_g + l * EE, bn2_b + l * EE);
    }

    // ---- output (fp32, B x N x E) ----
    (void)hipMemcpyAsync(d_out, xw, szXf, hipMemcpyDeviceToDevice, stream);
}
